// SelfAttention_87703232184998
// MI455X (gfx1250) — matrix operand order repaired and checked by reading, not compiled, not run on hardware
//
#include <hip/hip_runtime.h>

// ---------------------------------------------------------------------------
// Problem constants (from reference): B=8, C=512, L=2048, C8=64
// ---------------------------------------------------------------------------
#define BB  8
#define CC  512
#define LL  2048
#define C8  64
#define CM  640   // 64 (q) + 64 (k) + 512 (v) fused output rows

typedef __attribute__((ext_vector_type(16))) __bf16        v16bf;
typedef __attribute__((ext_vector_type(8)))  float         v8f;
typedef __attribute__((ext_vector_type(8)))  unsigned int  v8u;
typedef __attribute__((ext_vector_type(4)))  unsigned int  v4u;
typedef __attribute__((ext_vector_type(4)))  float         v4f;
// TDM descriptor group types (probe-verified builtin signature)
typedef __attribute__((ext_vector_type(4)))  unsigned int  tdm_g0_t;
typedef __attribute__((ext_vector_type(8)))  int           tdm_g1_t;
typedef __attribute__((ext_vector_type(4)))  int           tdm_g2_t;

__device__ __forceinline__ unsigned short f2bf(float f) {
    __bf16 h = (__bf16)f;                      // native v_cvt to bf16
    return __builtin_bit_cast(unsigned short, h);
}
__device__ __forceinline__ unsigned int pack2bf(float lo, float hi) {
    return (unsigned int)f2bf(lo) | ((unsigned int)f2bf(hi) << 16);
}
__device__ __forceinline__ v8f v8f_zero() {
    v8f z = {0.f,0.f,0.f,0.f,0.f,0.f,0.f,0.f};
    return z;
}
__device__ __forceinline__ v8f wmma_bf16(v8u a, v8u b, v8f c) {
    return __builtin_amdgcn_wmma_f32_16x16x32_bf16(
        false, __builtin_bit_cast(v16bf, a),
        false, __builtin_bit_cast(v16bf, b),
        (short)0, c, false, false);
}
// A-fragment (16x32, 16-bit) K-pair index for VGPR v, lane-half hi
__device__ __forceinline__ int a_kpair(int v, int hi) {
    return ((v < 4) ? (2 * v) : (16 + 2 * (v - 4))) + 8 * hi;
}
// B-fragment (32x16, 16-bit) K-pair index
__device__ __forceinline__ int b_kpair(int v, int hi) {
    return ((v < 4) ? (2 * v) : (16 + 2 * (v - 4))) + 8 * hi;
}

// ---------------------------------------------------------------------------
// Kernel 0: one-time f32 -> bf16 conversion of fused weights [Wq;Wk;Wv].
// 640x512 row-major bf16. Grid 160 x 256 threads, 8 elements/thread.
// ---------------------------------------------------------------------------
__global__ __launch_bounds__(256) void wcvt_kernel(
    const float* __restrict__ Wq, const float* __restrict__ Wk,
    const float* __restrict__ Wv, unsigned short* __restrict__ Wbf)
{
    const int t   = blockIdx.x * 256 + threadIdx.x;   // 0 .. 40959
    const int idx = t * 8;                            // element index
    const int o   = idx >> 9;                         // fused output row
    const int c   = idx & 511;
    const float* src;
    if      (o < 64)  src = Wq + (size_t)o * CC + c;
    else if (o < 128) src = Wk + (size_t)(o - 64) * CC + c;
    else              src = Wv + (size_t)(o - 128) * CC + c;
    const v4f a = *reinterpret_cast<const v4f*>(src);
    const v4f b = *reinterpret_cast<const v4f*>(src + 4);
    v4u pk;
    pk[0] = pack2bf(a[0], a[1]);
    pk[1] = pack2bf(a[2], a[3]);
    pk[2] = pack2bf(b[0], b[1]);
    pk[3] = pack2bf(b[2], b[3]);
    *reinterpret_cast<v4u*>(Wbf + idx) = pk;
}

// ---------------------------------------------------------------------------
// Kernel 1: fused 1x1-conv projections  Wbf (640x512 bf16) @ x[b] (512xL)
//   Q,K written as [B][L][64] bf16  (d contiguous -> 32-bit pair loads later)
//   V   written as [B][C][L]  bf16  (j contiguous -> P*V B-fragments)
// Grid: (L/32, B), 256 threads (8 waves). Wave w owns m-tiles {w, w+8,...}.
// ---------------------------------------------------------------------------
__global__ __launch_bounds__(256) void proj_kernel(
    const float* __restrict__ x,
    const unsigned short* __restrict__ Wbf,
    const float* __restrict__ bq, const float* __restrict__ bk,
    const float* __restrict__ bv,
    unsigned short* __restrict__ Qbf,
    unsigned short* __restrict__ Kbf,
    unsigned short* __restrict__ Vbf)
{
    __shared__ unsigned short xT[32 * 36];  // [l 0..31][c 0..31], stride 36 (pad)

    const int b     = blockIdx.y;
    const int lbase = blockIdx.x * 32;
    const int tid   = threadIdx.x;
    const int lane  = tid & 31;
    const int w     = tid >> 5;
    const int n16   = lane & 15;
    const int hi    = lane >> 4;
    const float* xb = x + (size_t)b * CC * LL;

    v8f acc[5][2];
    for (int mi = 0; mi < 5; ++mi)
        for (int nt = 0; nt < 2; ++nt) acc[mi][nt] = v8f_zero();

    for (int ks = 0; ks < 16; ++ks) {                 // K = 512 = 16 * 32
        const int kb = ks * 32;
        __syncthreads();
        // stage x tile transposed to bf16: xT[l][c], coalesced f32 reads
        {
            const int cc = tid >> 3;                  // 0..31
            const int l0 = (tid & 7) * 4;             // 0..28
            const float4 xv = *reinterpret_cast<const float4*>(
                xb + (size_t)(kb + cc) * LL + lbase + l0);
            xT[(l0 + 0) * 36 + cc] = f2bf(xv.x);
            xT[(l0 + 1) * 36 + cc] = f2bf(xv.y);
            xT[(l0 + 2) * 36 + cc] = f2bf(xv.z);
            xT[(l0 + 3) * 36 + cc] = f2bf(xv.w);
        }
        __syncthreads();
        // B fragments (x): K=c pairs contiguous in xT rows
        v8u bfr[2];
        for (int nt = 0; nt < 2; ++nt) {
            const int l = nt * 16 + n16;
            for (int v = 0; v < 8; ++v)
                bfr[nt][v] = *reinterpret_cast<const unsigned int*>(
                    &xT[l * 36 + b_kpair(v, hi)]);
        }
        // A fragments (pre-converted bf16 weights) + WMMA
        for (int mi = 0; mi < 5; ++mi) {
            const int mt   = w + 8 * mi;
            const int orow = mt * 16 + n16;
            const unsigned short* wrow = Wbf + (size_t)orow * CC + kb;
            v8u af;
            for (int v = 0; v < 8; ++v)
                af[v] = *reinterpret_cast<const unsigned int*>(
                    wrow + a_kpair(v, hi));
            acc[mi][0] = wmma_bf16(af, bfr[0], acc[mi][0]);
            acc[mi][1] = wmma_bf16(af, bfr[1], acc[mi][1]);
        }
    }

    // epilogue: bias + bf16 store. D-tile: M = obase + r (r=VGPR idx), N = n16.
    for (int mi = 0; mi < 5; ++mi) {
        const int mt    = w + 8 * mi;
        const int obase = mt * 16 + 8 * hi;           // multiple of 8
        for (int nt = 0; nt < 2; ++nt) {
            const int l = lbase + nt * 16 + n16;
            if (mt < 8) {                             // Q (mt<4) or K tile
                const float* bias = (mt < 4) ? bq : bk;
                const int o0 = obase - ((mt < 4) ? 0 : 64);
                v4u pk;
                for (int r = 0; r < 4; ++r)
                    pk[r] = pack2bf(acc[mi][nt][2 * r]     + bias[o0 + 2 * r],
                                    acc[mi][nt][2 * r + 1] + bias[o0 + 2 * r + 1]);
                unsigned short* dst = ((mt < 4) ? Qbf : Kbf)
                                    + ((size_t)b * LL + l) * C8 + o0;
                *reinterpret_cast<v4u*>(dst) = pk;    // 16B aligned
            } else {                                  // V tile -> [C][L]
                const int o0 = obase - 128;
                for (int r = 0; r < 8; ++r)
                    Vbf[((size_t)b * CC + o0 + r) * LL + l] =
                        f2bf(acc[mi][nt][r] + bv[o0 + r]);
            }
        }
    }
}

// ---------------------------------------------------------------------------
// Kernel 2: flash attention. Grid (L/32, B), 256 threads (8 waves).
//  Workgroup = 32 query positions; wave w owns output channels [64w, 64w+64).
//  K tile is staged LDS-side by the Tensor Data Mover (TENSOR_LOAD_TO_LDS)
//  with D#-programmed padding matching the bank-conflict-free 144B stride.
// ---------------------------------------------------------------------------
#define KT_STRIDE 72            // entries (144 B): 128B data + 16B TDM pad
#define KT_BYTES  (64 * KT_STRIDE * 2)          //  9216
#define VT_BYTES  (8 * 64 * KT_STRIDE * 2)      // 73728
#define SLD_BYTES (32 * 65 * 4)                 //  8320
#define PLD_BYTES (32 * KT_STRIDE * 2)          //  4608
#define RED_BYTES (32 * 8 * 4)                  //  1024
#define SM_TOTAL  (KT_BYTES + VT_BYTES + SLD_BYTES + PLD_BYTES + RED_BYTES + 3 * 32 * 4)

// Issue a TDM 2D tile load: global (rows x 32 dwords, stride 32 dwords)
// -> LDS at lds_byte_off with 4-dword padding after every 32 dwords.
__device__ __forceinline__ void tdm_load_k_tile(unsigned lds_byte_off,
                                                const void* gptr)
{
    const unsigned long long ga = (unsigned long long)(uintptr_t)gptr;
    tdm_g0_t g0;
    g0[0] = 1u;                                   // count=1, no gather
    g0[1] = lds_byte_off;                         // lds_addr
    g0[2] = (unsigned)(ga & 0xFFFFFFFFu);         // global_addr[31:0]
    g0[3] = (unsigned)((ga >> 32) & 0x01FFFFFFu)  // global_addr[56:32]
          | (2u << 30);                           // type = 2 ("image")
    tdm_g1_t g1;
    g1[0] = (int)((2u << 16)                      // data_size = 4 bytes
          | (1u << 20)                            // pad_enable
          | (4u << 22)                            // pad_interval: 32 dwords
          | (3u << 25));                          // pad_amount:   4 dwords
    g1[1] = (int)((32u & 0xFFFFu) << 16);         // tensor_dim0 = 32 (lo16)
    g1[2] = (int)((2048u & 0xFFFFu) << 16);       // dim0 hi=0 | tensor_dim1 lo
    g1[3] = (int)((2048u >> 16) | (32u << 16));   // tensor_dim1 hi | tile_dim0
    g1[4] = (int)64u;                             // tile_dim1=64, tile_dim2=0
    g1[5] = (int)32u;                             // tensor_dim0_stride = 32
    g1[6] = 0;                                    // stride hi | dim1_stride lo
    g1[7] = 0;
    tdm_g2_t gz = {0, 0, 0, 0};
#if __clang_major__ >= 23
    tdm_g1_t gz8 = {0,0,0,0,0,0,0,0};
    __builtin_amdgcn_tensor_load_to_lds(g0, g1, gz, gz, gz8, 0);
#else
    __builtin_amdgcn_tensor_load_to_lds(g0, g1, gz, gz, 0);
#endif
}

__global__ __launch_bounds__(256) void attn_kernel(
    const unsigned short* __restrict__ Qbf,
    const unsigned short* __restrict__ Kbf,
    const unsigned short* __restrict__ Vbf,
    const float* __restrict__ x,
    const float* __restrict__ gamma,
    float* __restrict__ out)
{
    extern __shared__ char smem[];
    unsigned short* Kt   = (unsigned short*)smem;
    unsigned short* Vt   = (unsigned short*)(smem + KT_BYTES);
    float*          Sld  = (float*)(smem + KT_BYTES + VT_BYTES);
    unsigned short* Pld  = (unsigned short*)(smem + KT_BYTES + VT_BYTES + SLD_BYTES);
    float*          red  = (float*)(smem + KT_BYTES + VT_BYTES + SLD_BYTES + PLD_BYTES);
    float*          mrow = red + 32 * 8;
    float*          lrow = mrow + 32;
    float*          rfac = lrow + 32;

    const int b     = blockIdx.y;
    const int ibase = blockIdx.x * 32;
    const int tid   = threadIdx.x;
    const int lane  = tid & 31;
    const int w     = tid >> 5;
    const int n16   = lane & 15;
    const int hi    = lane >> 4;
    const int cwave = w * 64;
    const unsigned ldsKt = __builtin_amdgcn_groupstaticsize();  // dyn-LDS base

    const unsigned short* Qb = Qbf + (size_t)b * LL * C8;
    const unsigned short* Kb = Kbf + (size_t)b * LL * C8;
    const unsigned short* Vb = Vbf + (size_t)b * CC * LL;

    // preload Q fragments (kept in VGPRs for whole kernel): qf[mtile][kstep]
    v8u qf[2][2];
    for (int mt = 0; mt < 2; ++mt)
        for (int ks = 0; ks < 2; ++ks)
            for (int v = 0; v < 8; ++v) {
                const int kk = ks * 32 + a_kpair(v, hi);
                qf[mt][ks][v] = *reinterpret_cast<const unsigned int*>(
                    Qb + (size_t)(ibase + mt * 16 + n16) * C8 + kk);
            }

    if (tid < 32) { mrow[tid] = -1e30f; lrow[tid] = 0.f; }
    v8f acc[2][4];
    for (int mt = 0; mt < 2; ++mt)
        for (int ct = 0; ct < 4; ++ct) acc[mt][ct] = v8f_zero();
    __syncthreads();

    for (int jt = 0; jt < LL / 64; ++jt) {
        const int jbase = jt * 64;
        // ---- K tile [64 j][64 d] via Tensor Data Mover (wave 0 only) ----
        if (w == 0)
            tdm_load_k_tile(ldsKt, Kb + (size_t)jbase * C8);
        // ---- stage this wave's V slice [64 c][64 j] ----
        {
            unsigned short* Vw = Vt + w * (64 * KT_STRIDE);
            for (int t = 0; t < 8; ++t) {
                const int sid = lane + 32 * t;
                const int ccr = sid >> 2, sg = sid & 3;
                const v4u vd = *reinterpret_cast<const v4u*>(
                    Vb + (size_t)(cwave + ccr) * LL + jbase + sg * 8);
                *reinterpret_cast<v4u*>(Vw + ccr * KT_STRIDE + sg * 8) = vd;
            }
        }
        if (jt + 1 < LL / 64) {  // prefetch next tiles (global_prefetch_b8)
            __builtin_prefetch(Kb + (size_t)(jbase + 64 + (tid >> 2)) * C8, 0, 0);
            __builtin_prefetch(Vb + (size_t)(cwave + (lane >> 1)) * LL + jbase + 64, 0, 0);
        }
        if (w == 0)
            __builtin_amdgcn_s_wait_tensorcnt(0);  // K tile landed in LDS
        __syncthreads();

        // ---- scores: wave w computes tile (mt = w>>2, jsub = w&3) ----
        {
            const int mt = w >> 2, jsub = w & 3;
            v8f s = v8f_zero();
            for (int ks = 0; ks < 2; ++ks) {           // d = 64 = 2*32
                v8u kf;
                for (int v = 0; v < 8; ++v) {
                    const int dd = ks * 32 + b_kpair(v, hi);
                    kf[v] = *reinterpret_cast<const unsigned int*>(
                        &Kt[(jsub * 16 + n16) * KT_STRIDE + dd]);
                }
                s = wmma_bf16(qf[mt][ks], kf, s);
            }
            for (int r = 0; r < 8; ++r)                // D: M=8*hi+r, N=n16
                Sld[(mt * 16 + 8 * hi + r) * 65 + jsub * 16 + n16] = s[r];
        }
        __syncthreads();

        // ---- online softmax: 8 threads per row, 8 keys each ----
        const int rr = tid >> 3, sseg = tid & 7;
        {
            float m = -1e30f;
            for (int jj = 0; jj < 8; ++jj)
                m = fmaxf(m, Sld[rr * 65 + sseg * 8 + jj]);
            red[rr * 8 + sseg] = m;
        }
        __syncthreads();
        if (tid < 32) {
            float mx = mrow[tid];
            for (int s2 = 0; s2 < 8; ++s2) mx = fmaxf(mx, red[tid * 8 + s2]);
            rfac[tid] = __expf(mrow[tid] - mx);
            mrow[tid] = mx;
        }
        __syncthreads();
        {
            const float mx = mrow[rr];
            float ps = 0.f;
            for (int jj = 0; jj < 8; ++jj) {
                const float p = __expf(Sld[rr * 65 + sseg * 8 + jj] - mx);
                ps += p;
                Pld[rr * KT_STRIDE + sseg * 8 + jj] = f2bf(p);
            }
            red[rr * 8 + sseg] = ps;
        }
        __syncthreads();
        if (tid < 32) {
            float s = 0.f;
            for (int s2 = 0; s2 < 8; ++s2) s += red[tid * 8 + s2];
            lrow[tid] = lrow[tid] * rfac[tid] + s;
        }

        // ---- rescale accumulators, then P*V ----
        float rf[2][8];
        for (int mt = 0; mt < 2; ++mt)
            for (int r = 0; r < 8; ++r)
                rf[mt][r] = rfac[mt * 16 + 8 * hi + r];
        for (int mt = 0; mt < 2; ++mt)
            for (int ct = 0; ct < 4; ++ct)
                for (int r = 0; r < 8; ++r)
                    acc[mt][ct][r] *= rf[mt][r];

        unsigned short* Vw = Vt + w * (64 * KT_STRIDE);
        for (int ks = 0; ks < 2; ++ks) {               // j = 64 = 2*32
            v8u pf[2];
            for (int mt = 0; mt < 2; ++mt)
                for (int v = 0; v < 8; ++v) {
                    const int kk = ks * 32 + a_kpair(v, hi);
                    pf[mt][v] = *reinterpret_cast<const unsigned int*>(
                        &Pld[(mt * 16 + n16) * KT_STRIDE + kk]);
                }
            for (int ct = 0; ct < 4; ++ct) {
                v8u vf;
                for (int v = 0; v < 8; ++v) {
                    const int kk = ks * 32 + b_kpair(v, hi);
                    vf[v] = *reinterpret_cast<const unsigned int*>(
                        &Vw[(ct * 16 + n16) * KT_STRIDE + kk]);
                }
                acc[0][ct] = wmma_bf16(pf[0], vf, acc[0][ct]);
                acc[1][ct] = wmma_bf16(pf[1], vf, acc[1][ct]);
            }
        }
        __syncthreads();
    }

    // ---- epilogue: out = gamma * acc / l + x, 128-bit I/O (i innermost) ----
    const float g = gamma[0];
    float linv[2][8];
    for (int mt = 0; mt < 2; ++mt)
        for (int r = 0; r < 8; ++r)
            linv[mt][r] = 1.0f / lrow[mt * 16 + 8 * hi + r];
    for (int mt = 0; mt < 2; ++mt)
        for (int ct = 0; ct < 4; ++ct) {
            const int c = cwave + ct * 16 + n16;       // D: N = n16 -> channel
            const size_t base = ((size_t)b * CC + c) * LL + ibase + mt * 16 + 8 * hi;
            const v4f x0 = *reinterpret_cast<const v4f*>(x + base);
            const v4f x1 = *reinterpret_cast<const v4f*>(x + base + 4);
            v4f o0, o1;
            for (int r = 0; r < 4; ++r) {
                o0[r] = g * acc[mt][ct][r]     * linv[mt][r]     + x0[r];
                o1[r] = g * acc[mt][ct][r + 4] * linv[mt][r + 4] + x1[r];
            }
            *reinterpret_cast<v4f*>(out + base)     = o0;
            *reinterpret_cast<v4f*>(out + base + 4) = o1;
        }
}

// ---------------------------------------------------------------------------
extern "C" void kernel_launch(void* const* d_in, const int* in_sizes, int n_in,
                              void* d_out, int out_size, void* d_ws, size_t ws_size,
                              hipStream_t stream) {
    (void)in_sizes; (void)n_in; (void)out_size; (void)ws_size;
    const float* x     = (const float*)d_in[0];
    const float* Wq    = (const float*)d_in[1];
    const float* bq    = (const float*)d_in[2];
    const float* Wk    = (const float*)d_in[3];
    const float* bk    = (const float*)d_in[4];
    const float* Wv    = (const float*)d_in[5];
    const float* bv    = (const float*)d_in[6];
    const float* gamma = (const float*)d_in[7];
    float* out = (float*)d_out;

    unsigned short* Qbf = (unsigned short*)d_ws;                 //  2 MB
    unsigned short* Kbf = Qbf + (size_t)BB * LL * C8;            //  2 MB
    unsigned short* Vbf = Kbf + (size_t)BB * LL * C8;            // 16 MB
    unsigned short* Wbf = Vbf + (size_t)BB * CC * LL;            // 640 KB

    dim3 blk(256);
    wcvt_kernel<<<dim3(CM * CC / (256 * 8)), blk, 0, stream>>>(Wq, Wk, Wv, Wbf);
    dim3 grid(LL / 32, BB);
    proj_kernel<<<grid, blk, 0, stream>>>(x, Wbf, bq, bk, bv, Qbf, Kbf, Vbf);
    attn_kernel<<<grid, blk, SM_TOTAL, stream>>>(Qbf, Kbf, Vbf, x, gamma, out);
}
